// LearnForces2_43104291783447
// MI455X (gfx1250) — compile-verified, matrix operand order checked by reading, not run on hardware
//
#include <hip/hip_runtime.h>

typedef __attribute__((ext_vector_type(16))) _Float16 v16h;
typedef __attribute__((ext_vector_type(8)))  float    v8f;
typedef __attribute__((ext_vector_type(2)))  _Float16 h2v;
typedef unsigned int u32;

#define NP     10
#define NE     45
#define NT     100000
#define TPW    32                     // timesteps per wave
#define WAVES  8
#define CHUNKS 45                     // (TPW*NE)/32
#define NWAVE  ((NT + TPW - 1) / TPW) // 3125
#define RSTR   17                     // padded row stride (u32) for staging buffers

__device__ __forceinline__ void lds_fence() {
    asm volatile("s_wait_dscnt 0" ::: "memory");
    __builtin_amdgcn_wave_barrier();
}

__device__ __forceinline__ float fast_tanh(float x) {
    // tanh(x) = (e^{2x}-1)/(e^{2x}+1); clamp so exp2 never overflows
    x = fminf(fmaxf(x, -9.0f), 9.0f);
    float e = __builtin_amdgcn_exp2f(x * 2.885390081777927f); // 2*log2(e)
    return (e - 1.0f) * __builtin_amdgcn_rcpf(e + 1.0f);
}

__global__ __launch_bounds__(256)
void lf2_kernel(const float* __restrict__ D_V,  const float* __restrict__ logm,
                const float* __restrict__ W1,   const float* __restrict__ b1,
                const float* __restrict__ W2,   const float* __restrict__ b2,
                const float* __restrict__ W3,   const float* __restrict__ b3,
                const int*  __restrict__ snd_g, const int* __restrict__ rcv_g,
                float* __restrict__ out)
{
    __shared__ float sLM[NP];
    __shared__ float sIM[NP];
    __shared__ int   sSnd[NE];
    __shared__ int   sRcv[NE];
    __shared__ u32   featbuf[WAVES][32 * 4];        // 4 KB : f16 features, 4 u32 (8 halfs)/row
    __shared__ u32   h1buf  [WAVES][16 * RSTR];     // ~8.7 KB : layer-1 acts, permuted-K pairs
    __shared__ u32   h2buf  [WAVES][16 * RSTR];     // ~8.7 KB : layer-2 acts, permuted-K pairs
    __shared__ float esph   [WAVES][32 * 3];        // 3 KB : layer-3 outputs per chunk
    __shared__ float acc    [WAVES][TPW * NP * 3];  // 30 KB : per-wave force accumulators

    const int tid = threadIdx.x;
    if (tid < NP) {
        float v = fminf(fmaxf(logm[tid], -12.0f), 12.0f);
        sLM[tid] = v;
        sIM[tid] = __expf(-v);
    }
    if (tid < NE) { sSnd[tid] = snd_g[tid]; sRcv[tid] = rcv_g[tid]; }
    __syncthreads();

    const int w    = tid >> 5;
    const int lane = tid & 31;
    const int wid  = blockIdx.x * WAVES + w;
    if (wid >= NWAVE) return;   // 3125 waves exactly cover 100000 timesteps

    for (int i = lane; i < TPW * NP * 3; i += 32) acc[w][i] = 0.0f;

    // CDNA5 16-bit 16x32 A-fragment K map: lane pair index -> K positions
    const int hsel = lane >> 4;
    const int nl   = lane & 15;
    int K0v[8], Kh[8];
#pragma unroll
    for (int v = 0; v < 8; ++v) {
        K0v[v] = ((v < 4) ? 2 * v : 16 + 2 * (v - 4)) + 8 * hsel;  // even
        Kh[v]  = K0v[v] >> 1;  // u32 index within a staged row; also k_orig low half
    }

    union VU { v16h v; _Float16 e[16]; u32 u[8]; };

    // ---- Build B fragments once (straight from global memory) ----
    // Layer 1: identity K ordering, K=0..4 valid (features), rest zero-padded.
    // Layers 2/3: permuted-K ordering k_orig(p) = (p even) ? p/2 : 16 + p/2, so the
    // tanh'd C fragments can be stored as packed u32 pairs (cols n and n+16 adjacent).
    VU bw1[2], bw2[2], bw3;
#pragma unroll
    for (int n = 0; n < 2; ++n) {
        int N = n * 16 + nl;
#pragma unroll
        for (int v = 0; v < 8; ++v) {
            int K0 = K0v[v];
            bw1[n].e[2 * v]     = (K0 < 5)     ? (_Float16)W1[K0 * 32 + N]       : (_Float16)0.0f;
            bw1[n].e[2 * v + 1] = (K0 + 1 < 5) ? (_Float16)W1[(K0 + 1) * 32 + N] : (_Float16)0.0f;
            int k0 = Kh[v];
            bw2[n].e[2 * v]     = (_Float16)W2[k0 * 32 + N];
            bw2[n].e[2 * v + 1] = (_Float16)W2[(16 + k0) * 32 + N];
        }
    }
    {
        bool valid = nl < 3;
#pragma unroll
        for (int v = 0; v < 8; ++v) {
            int k0 = Kh[v];
            bw3.e[2 * v]     = valid ? (_Float16)W3[k0 * 3 + nl]        : (_Float16)0.0f;
            bw3.e[2 * v + 1] = valid ? (_Float16)W3[(16 + k0) * 3 + nl] : (_Float16)0.0f;
        }
    }

    const float b1a = b1[nl],      b1b = b1[16 + nl];
    const float b2a = b2[nl],      b2b = b2[16 + nl];
    const float b3v = (nl < 3) ? b3[nl] : 0.0f;

    const long ebase = (long)wid * (TPW * NE);
    lds_fence();

    for (int c = 0; c < CHUNKS; ++c) {
        // ---------- Phase A: lane-per-edge front end ----------
        int  idx = c * 32 + lane;          // 0..1439 within this wave
        int  tl  = idx / NE;               // local timestep
        int  le  = idx - tl * NE;          // edge within timestep
        long eg  = ebase + idx;

        float x = D_V[eg * 3 + 0], y = D_V[eg * 3 + 1], z = D_V[eg * 3 + 2];
        float r  = sqrtf(x * x + y * y + z * z);
        float ct = fminf(fmaxf(z / r, -1.0f), 1.0f);
        float th = acosf(ct);
        float ph = atan2f(y, x);

        int   snd = sSnd[le], rcv = sRcv[le];
        float f3  = sLM[rcv], f4 = sLM[snd];

        // stage features: [r, th, ph, f3, f4, 0, 0, 0] as f16 -> 16B store
        union PK { h2v h; u32 u; } p0, p1, p2;
        p0.h.x = (_Float16)r;  p0.h.y = (_Float16)th;
        p1.h.x = (_Float16)ph; p1.h.y = (_Float16)f3;
        p2.h.x = (_Float16)f4; p2.h.y = (_Float16)0.0f;
        int4* fq = (int4*)&featbuf[w][lane * 4];
        *fq = make_int4((int)p0.u, (int)p1.u, (int)p2.u, 0);
        lds_fence();

        // ---------- Phase B: WMMA layers 1,2,3 for two 16-edge tiles ----------
        const u32* fbr = (const u32*)&featbuf[w][0];
#pragma unroll
        for (int tt = 0; tt < 2; ++tt) {
            // layer 1: A = features (K=0..7 real, rest zero)
            VU a1;
            int arow = tt * 16 + nl;
#pragma unroll
            for (int v = 0; v < 4; ++v) {
                u32 t = fbr[arow * 4 + v];
                a1.u[v] = hsel ? 0u : t;   // lanes 16-31 cover K=8..15/24..31 -> padded zero
            }
#pragma unroll
            for (int v = 4; v < 8; ++v) a1.u[v] = 0u;

            v8f c0, c1;
#pragma unroll
            for (int v = 0; v < 8; ++v) { c0[v] = b1a; c1[v] = b1b; }
            c0 = __builtin_amdgcn_wmma_f32_16x16x32_f16(false, a1.v, false, bw1[0].v,
                                                        (short)0, c0, false, false);
            c1 = __builtin_amdgcn_wmma_f32_16x16x32_f16(false, a1.v, false, bw1[1].v,
                                                        (short)0, c1, false, false);
            // tanh + packed store (permuted-K: cols nl, nl+16 -> adjacent halfs)
            u32* h1p = &h1buf[w][0];
#pragma unroll
            for (int v = 0; v < 8; ++v) {
                int row = v + 8 * hsel;                  // C layout: M = v + 8*(lane/16)
                union PK pk;
                pk.h.x = (_Float16)fast_tanh(c0[v]);
                pk.h.y = (_Float16)fast_tanh(c1[v]);
                h1p[row * RSTR + nl] = pk.u;
            }
            lds_fence();

            // layer 2
            VU a2;
            const u32* h1r = &h1buf[w][0];
#pragma unroll
            for (int v = 0; v < 8; ++v) a2.u[v] = h1r[nl * RSTR + Kh[v]];
            v8f d0, d1;
#pragma unroll
            for (int v = 0; v < 8; ++v) { d0[v] = b2a; d1[v] = b2b; }
            d0 = __builtin_amdgcn_wmma_f32_16x16x32_f16(false, a2.v, false, bw2[0].v,
                                                        (short)0, d0, false, false);
            d1 = __builtin_amdgcn_wmma_f32_16x16x32_f16(false, a2.v, false, bw2[1].v,
                                                        (short)0, d1, false, false);
            u32* h2p = &h2buf[w][0];
#pragma unroll
            for (int v = 0; v < 8; ++v) {
                int row = v + 8 * hsel;
                union PK pk;
                pk.h.x = (_Float16)fast_tanh(d0[v]);
                pk.h.y = (_Float16)fast_tanh(d1[v]);
                h2p[row * RSTR + nl] = pk.u;
            }
            lds_fence();

            // layer 3 (W3 zero-padded to 32x16; only N=0..2 meaningful)
            VU a3;
            const u32* h2r = &h2buf[w][0];
#pragma unroll
            for (int v = 0; v < 8; ++v) a3.u[v] = h2r[nl * RSTR + Kh[v]];
            v8f c2;
#pragma unroll
            for (int v = 0; v < 8; ++v) c2[v] = b3v;
            c2 = __builtin_amdgcn_wmma_f32_16x16x32_f16(false, a3.v, false, bw3.v,
                                                        (short)0, c2, false, false);
            if (nl < 3) {
#pragma unroll
                for (int v = 0; v < 8; ++v) {
                    int row = v + 8 * hsel;
                    esph[w][(tt * 16 + row) * 3 + nl] = c2[v];
                }
            }
            lds_fence();
        }

        // ---------- Phase C: sph->cart + +/- scatter into per-wave accumulator ----------
        float er  = esph[w][lane * 3 + 0];
        float eth = esph[w][lane * 3 + 1];
        float eph = esph[w][lane * 3 + 2];
        float st, ctn, sp, cp;
        __sincosf(eth, &st, &ctn);
        __sincosf(eph, &sp, &cp);
        float cx = er * st * cp, cy = er * st * sp, cz = er * ctn;

        float* ar = &acc[w][(tl * NP + rcv) * 3];
        atomicAdd(&ar[0],  cx); atomicAdd(&ar[1],  cy); atomicAdd(&ar[2],  cz);
        float* as = &acc[w][(tl * NP + snd) * 3];
        atomicAdd(&as[0], -cx); atomicAdd(&as[1], -cy); atomicAdd(&as[2], -cz);
        lds_fence();
    }

    lds_fence();
    // accel = forces * exp(-clip(logm)); acc index i == t*30 + p*3 + comp
    for (int i = lane; i < TPW * NP * 3; i += 32) {
        int t   = i / 30;
        int rem = i - t * 30;
        int p   = rem / 3;
        out[(long)(wid * TPW + t) * 30 + rem] = acc[w][i] * sIM[p];
    }
}

extern "C" void kernel_launch(void* const* d_in, const int* in_sizes, int n_in,
                              void* d_out, int out_size, void* d_ws, size_t ws_size,
                              hipStream_t stream) {
    (void)in_sizes; (void)n_in; (void)out_size; (void)d_ws; (void)ws_size;
    const float* D_V  = (const float*)d_in[0];
    const float* logm = (const float*)d_in[1];
    const float* W1   = (const float*)d_in[2];
    const float* b1   = (const float*)d_in[3];
    const float* W2   = (const float*)d_in[4];
    const float* b2   = (const float*)d_in[5];
    const float* W3   = (const float*)d_in[6];
    const float* b3   = (const float*)d_in[7];
    const int*   snd  = (const int*)d_in[8];
    const int*   rcv  = (const int*)d_in[9];
    float* out = (float*)d_out;

    const int blocks = (NWAVE + WAVES - 1) / WAVES;  // 391
    lf2_kernel<<<blocks, 256, 0, stream>>>(D_V, logm, W1, b1, W2, b2, W3, b3,
                                           snd, rcv, out);
}
